// ConditionalLinearAttention_61014305407449
// MI455X (gfx1250) — matrix operand order repaired and checked by reading, not compiled, not run on hardware
//
#include <hip/hip_runtime.h>
#include <hip/hip_bf16.h>

// MI455X / gfx1250, wave32. WMMA f32 <- f16 x f16, 16x16x32.
typedef __attribute__((ext_vector_type(16))) _Float16 v16h;
typedef __attribute__((ext_vector_type(8)))  float    v8f;
typedef __attribute__((ext_vector_type(4)))  int      v4i;
typedef __attribute__((address_space(1))) v4i gv4i;   // global int4
typedef __attribute__((address_space(3))) v4i lv4i;   // LDS int4

static __device__ __forceinline__ v8f wmma_f16(v16h a, v16h b, v8f c) {
  // (neg_a, A, neg_b, B, c_mod, C, reuse_a, reuse_b)
  return __builtin_amdgcn_wmma_f32_16x16x32_f16(false, a, false, b, (short)0, c, false, false);
}

static __device__ __forceinline__ int lane_id() { return (int)(threadIdx.x & 31u); }

// ---- WMMA fragment loaders (contiguous k order, k = 16*h + s, same in A and B) -------------------------
// A (16x32, MxK): lane<16 -> row m0+lane, K = k0..k0+15
//                 lane>=16 -> row m0+lane-16, K = k0+16..k0+31.
static __device__ __forceinline__ v16h load_a_f32(const float* A, int ld, int m0, int k0) {
  const int lane = lane_id();
  const float* p = A + (m0 + (lane & 15)) * ld + k0 + ((lane >> 4) << 4);
  v16h a;
#pragma unroll
  for (int i = 0; i < 8; ++i) { a[i] = (_Float16)p[i]; a[i + 8] = (_Float16)p[8 + i]; }
  return a;
}
static __device__ __forceinline__ v16h load_a_f16(const _Float16* A, int ld, int m0, int k0) {
  const int lane = lane_id();
  const _Float16* p = A + (m0 + (lane & 15)) * ld + k0 + ((lane >> 4) << 4);
  v16h a;
#pragma unroll
  for (int i = 0; i < 8; ++i) { a[i] = p[i]; a[i + 8] = p[8 + i]; }
  return a;
}
// B (32x16, KxN) from row-major KxN source: lane<16 -> col n0+lane, K=k0..k0+15;
// lanes 16..31 -> same cols, K=k0+16..k0+31.
static __device__ __forceinline__ v16h load_b_f32(const float* B, int ld, int k0, int n0) {
  const int lane = lane_id();
  const float* p = B + (k0 + ((lane >> 4) << 4)) * ld + n0 + (lane & 15);
  v16h b;
#pragma unroll
  for (int i = 0; i < 16; ++i) b[i] = (_Float16)p[i * ld];
  return b;
}
// B (KxN) from row-major N x K source (i.e. B[k][n] = S[n][k])
static __device__ __forceinline__ v16h load_bt_f32(const float* S, int ld, int k0, int n0) {
  const int lane = lane_id();
  const float* p = S + (n0 + (lane & 15)) * ld + k0 + ((lane >> 4) << 4);
  v16h b;
#pragma unroll
  for (int i = 0; i < 16; ++i) b[i] = (_Float16)p[i];
  return b;
}
static __device__ __forceinline__ v16h load_bt_f16(const _Float16* S, int ld, int k0, int n0) {
  const int lane = lane_id();
  const _Float16* p = S + (n0 + (lane & 15)) * ld + k0 + ((lane >> 4) << 4);
  v16h b;
#pragma unroll
  for (int i = 0; i < 16; ++i) b[i] = p[i];
  return b;
}
// C/D (16x16 f32, 8 VGPRs): VGPR r -> row m0+r (lanes 0..15) / m0+8+r (lanes 16..31)
static __device__ __forceinline__ void store_tile_f32(float* D, int ld, int m0, int n0,
                                                      v8f acc, const float* bias) {
  const int lane = lane_id();
  const int col = n0 + (lane & 15);
  const int rb  = m0 + ((lane >> 4) << 3);
#pragma unroll
  for (int r = 0; r < 8; ++r) {
    float v = acc[r];
    if (bias) v += bias[rb + r];
    D[(rb + r) * ld + col] = v;
  }
}
static __device__ __forceinline__ void store_tile_f16(_Float16* D, int ld, int m0, int n0, v8f acc) {
  const int lane = lane_id();
  const int col = n0 + (lane & 15);
  const int rb  = m0 + ((lane >> 4) << 3);
#pragma unroll
  for (int r = 0; r < 8; ++r) D[(rb + r) * ld + col] = (_Float16)acc[r];
}

// =========================================================================
// Kernel 1: per (batch, head): kv GEMM -> softmax(k) -> context -> M1 slice
//   M1[b][o][h*64+d] = sum_e Wout[o][h*64+e] * context[d][e]
// =========================================================================
__global__ __launch_bounds__(256) void k1_context_m1(
    const float* __restrict__ cond,    // [8][512][128]
    const float* __restrict__ Wcond,   // [1024][512]
    const float* __restrict__ Wout,    // [256][512]
    _Float16* __restrict__ M1)         // [8][256][512] f16
{
  __shared__ float    k_lds[64 * 128];  // 32 KB, pre/post softmax k
  __shared__ _Float16 v_lds[64 * 128];  // 16 KB
  __shared__ float    c_lds[64 * 64];   // 16 KB, context

  const int h = blockIdx.x;
  const int b = blockIdx.y;
  const int wave = (int)(threadIdx.x >> 5);
  const float* cf = cond + (size_t)b * 512 * 128;

  // ---- Phase A: k = Wcond[h*64 .. +64) @ cf ; v = Wcond[512+h*64 ..) @ cf
  {
    const int n0 = wave * 16;                       // each wave owns one N-subtile
    v8f z = {};
    v8f accK[4], accV[4];
#pragma unroll
    for (int m = 0; m < 4; ++m) { accK[m] = z; accV[m] = z; }
    const float* Ak = Wcond + (size_t)(h * 64) * 512;
    const float* Av = Wcond + (size_t)(512 + h * 64) * 512;
    for (int kt = 0; kt < 16; ++kt) {               // K = 512
      v16h bf = load_b_f32(cf, 128, kt * 32, n0);
#pragma unroll
      for (int m = 0; m < 4; ++m) {
        v16h a0 = load_a_f32(Ak, 512, m * 16, kt * 32);
        accK[m] = wmma_f16(a0, bf, accK[m]);
        v16h a1 = load_a_f32(Av, 512, m * 16, kt * 32);
        accV[m] = wmma_f16(a1, bf, accV[m]);
      }
    }
#pragma unroll
    for (int m = 0; m < 4; ++m) {
      store_tile_f32(k_lds, 128, m * 16, n0, accK[m], nullptr);
      store_tile_f16(v_lds, 128, m * 16, n0, accV[m]);
    }
  }
  __syncthreads();

  // ---- Phase B: softmax along m (128) per row of k
  if (threadIdx.x < 64) {
    float* row = k_lds + threadIdx.x * 128;
    float mx = -3.4e38f;
    for (int j = 0; j < 128; ++j) mx = fmaxf(mx, row[j]);
    float s = 0.f;
    for (int j = 0; j < 128; ++j) { float e = __expf(row[j] - mx); row[j] = e; s += e; }
    float inv = 1.f / s;
    for (int j = 0; j < 128; ++j) row[j] *= inv;
  }
  __syncthreads();

  // ---- Phase C: context[d][e] = sum_m k[d][m] * v[e][m]  (64x64, K=128)
  for (int t = wave * 2; t < wave * 2 + 2; ++t) {
    const int m0 = (t >> 2) * 16, n0 = (t & 3) * 16;
    v8f acc = {};
#pragma unroll
    for (int kt = 0; kt < 4; ++kt) {
      v16h a  = load_a_f32(k_lds, 128, m0, kt * 32);
      v16h bb = load_bt_f16(v_lds, 128, kt * 32, n0);
      acc = wmma_f16(a, bb, acc);
    }
    store_tile_f32(c_lds, 64, m0, n0, acc, nullptr);
  }
  __syncthreads();

  // ---- Phase D: M1 slice (256x64) = Wout[:, h*64:+64] @ contextT  (K=64)
  _Float16* M1b = M1 + (size_t)b * 256 * 512;
  const float* WoutH = Wout + h * 64;               // ld = 512
  for (int t = wave * 8; t < wave * 8 + 8; ++t) {
    const int m0 = (t >> 2) * 16, n0 = (t & 3) * 16;
    v8f acc = {};
#pragma unroll
    for (int kt = 0; kt < 2; ++kt) {
      v16h a  = load_a_f32(WoutH, 512, m0, kt * 32);
      v16h bb = load_bt_f32(c_lds, 64, kt * 32, n0); // B[e][d] = C[d][e]
      acc = wmma_f16(a, bb, acc);
    }
    store_tile_f16(M1b, 512, m0, h * 64 + n0, acc);
  }
}

// =========================================================================
// Kernel 2: W_eff[b] (256x256) = M1[b] (256x512) @ Wq (512x256)
// =========================================================================
__global__ __launch_bounds__(256) void k2_weff(
    const _Float16* __restrict__ M1,   // [8][256][512]
    const float* __restrict__ Wq,      // [512][256]
    _Float16* __restrict__ Weff)       // [8][256][256]
{
  const int mquad = blockIdx.x;        // 0..3: quarter of the M rows
  const int b     = blockIdx.y;
  const int wave  = (int)(threadIdx.x >> 5);
  const _Float16* M1b = M1 + (size_t)b * 256 * 512;
  _Float16* Wb = Weff + (size_t)b * 256 * 256;

  for (int half = 0; half < 2; ++half) {
    const int n0 = (wave + half * 8) * 16;
    v16h bfr[16];                                    // all-K Wq slice, 128 VGPRs
#pragma unroll
    for (int kt = 0; kt < 16; ++kt) bfr[kt] = load_b_f32(Wq, 256, kt * 32, n0);
    for (int mi = 0; mi < 4; ++mi) {
      const int m0 = (mquad * 4 + mi) * 16;
      v8f acc = {};
#pragma unroll
      for (int kt = 0; kt < 16; ++kt) {
        v16h a = load_a_f16(M1b, 512, m0, kt * 32);
        acc = wmma_f16(a, bfr[kt], acc);
      }
      store_tile_f16(Wb, 256, m0, n0, acc);
    }
  }
}

// =========================================================================
// Kernel 3 (memory-bound main pass):
//   out[b] (256x4096) = W_eff[b] @ x[b] + b_out       traffic ~67 MB total
// W_eff[b] (f16, 128 KB) is staged once per block into LDS via the gfx1250
// async global->LDS path (ASYNCcnt), then A-fragments come from ds_load.
// LDS rows padded to 264 halfs (132 dwords; 132 mod 64 = 4) so the 16 rows
// of an A-fragment land in 16 distinct banks.
// =========================================================================
#define WLD 264   // padded LDS leading dim (halfs), 16B-aligned row stride

__global__ __launch_bounds__(256) void k3_out(
    const _Float16* __restrict__ Weff, // [8][256][256], L2-resident
    const float* __restrict__ x,       // [8][256][4096]
    const float* __restrict__ bout,    // [256]
    float* __restrict__ out)           // [8][256][4096]
{
  __shared__ _Float16 w_lds[256 * WLD]; // 132 KB

  const int b   = blockIdx.y;
  const int tid = (int)threadIdx.x;
  const int n0  = blockIdx.x * 128 + (tid >> 5) * 16;
  const float* xb = x + (size_t)b * 256 * 4096;
  float* ob = out + (size_t)b * 256 * 4096;
  const _Float16* Wb = Weff + (size_t)b * 256 * 256;

  // ---- Stage W_eff[b] (256 rows x 512 B) into padded LDS, 16 B per chunk.
  // chunk c (0..8191): row = c>>5, col16 = c&31; each thread moves 32 chunks.
#if __has_builtin(__builtin_amdgcn_global_load_async_to_lds_b128)
  {
#pragma unroll
    for (int i = 0; i < 32; ++i) {
      const int c   = tid + i * 256;
      const int row = c >> 5, col = c & 31;
      __builtin_amdgcn_global_load_async_to_lds_b128(
          (gv4i*)((const char*)Wb + (size_t)c * 16),
          (lv4i*)((char*)w_lds + (size_t)row * (WLD * 2) + col * 16),
          0, 0);
    }
#if __has_builtin(__builtin_amdgcn_s_wait_asynccnt)
    __builtin_amdgcn_s_wait_asynccnt(0);
#else
    asm volatile("s_wait_asynccnt 0x0" ::: "memory");
#endif
  }
#else
  {
    for (int i = 0; i < 32; ++i) {
      const int c   = tid + i * 256;
      const int row = c >> 5, col = c & 31;
      *(v4i*)((char*)w_lds + (size_t)row * (WLD * 2) + col * 16) =
          *(const v4i*)((const char*)Wb + (size_t)c * 16);
    }
  }
#endif
  __syncthreads();

  // ---- Hold the full K=256 x-slice for this wave's 16 columns in registers.
  v16h bfr[8];
#pragma unroll
  for (int kt = 0; kt < 8; ++kt) bfr[kt] = load_b_f32(xb, 4096, kt * 32, n0);

  for (int mt = 0; mt < 16; ++mt) {
    v8f acc = {};
#pragma unroll
    for (int kt = 0; kt < 8; ++kt) {
      v16h a = load_a_f16(w_lds, WLD, mt * 16, kt * 32);
      acc = wmma_f16(a, bfr[kt], acc);
    }
    store_tile_f32(ob, 4096, mt * 16, n0, acc, bout);
  }
}

// =========================================================================
extern "C" void kernel_launch(void* const* d_in, const int* in_sizes, int n_in,
                              void* d_out, int out_size, void* d_ws, size_t ws_size,
                              hipStream_t stream) {
  const float* x     = (const float*)d_in[0];   // (8,256,64,64)
  const float* cond  = (const float*)d_in[1];   // (8,512,1,128)
  const float* Wq    = (const float*)d_in[2];   // (512,256)
  const float* Wcond = (const float*)d_in[3];   // (1024,512)
  const float* Wout  = (const float*)d_in[4];   // (256,512)
  const float* bout  = (const float*)d_in[5];   // (256,)
  float* out = (float*)d_out;

  // Workspace layout: M1 f16 [8][256][512] (2 MB) | Weff f16 [8][256][256] (1 MB)
  _Float16* M1   = (_Float16*)d_ws;
  _Float16* Weff = (_Float16*)((char*)d_ws + (size_t)8 * 256 * 512 * sizeof(_Float16));

  k1_context_m1<<<dim3(8, 8), 256, 0, stream>>>(cond, Wcond, Wout, M1);
  k2_weff<<<dim3(4, 8), 256, 0, stream>>>(M1, Wq, Weff);
  k3_out<<<dim3(32, 8), 256, 0, stream>>>(Weff, x, bout, out);
}
